// ReversibleTransformerEncoderLayer_30889404793236
// MI455X (gfx1250) — matrix operand order repaired and checked by reading, not compiled, not run on hardware
//
#include <hip/hip_runtime.h>
#include <hip/hip_bf16.h>
#include <cstdint>

#define TT   1024
#define BB   8
#define DDIM 1024
#define HH   16
#define DH   64
#define FF   4096
#define D3   3072

typedef __attribute__((ext_vector_type(16))) __bf16 v16bf;
typedef __attribute__((ext_vector_type(8)))  float  v8f;
typedef __attribute__((ext_vector_type(4)))  int    v4i;

union Frag { v16bf v; uint4 q[2]; };

__device__ __forceinline__ unsigned short f2bf(float f) {
  unsigned int u = __builtin_bit_cast(unsigned int, f);
  u += 0x7FFFu + ((u >> 16) & 1u);
  return (unsigned short)(u >> 16);
}
__device__ __forceinline__ float bf2f(unsigned short h) {
  unsigned int u = ((unsigned int)h) << 16;
  return __builtin_bit_cast(float, u);
}
__device__ __forceinline__ v8f wmma_bf(const Frag& a, const Frag& b, v8f c) {
  return __builtin_amdgcn_wmma_f32_16x16x32_bf16(false, a.v, false, b.v,
                                                 (short)0, c, false, false);
}

// Async 16-byte global -> LDS copy (per-lane), tracked by ASYNCcnt.
__device__ __forceinline__ void async_cp16(const unsigned short* g,
                                           unsigned short* l) {
  __builtin_amdgcn_global_load_async_to_lds_b128(
      (__attribute__((address_space(1))) v4i*)g,
      (__attribute__((address_space(3))) v4i*)l, 0, 0);
}

// ---------------- elementwise f32 -> bf16 convert ----------------
__global__ __launch_bounds__(256) void cvt_bf16_kernel(
    const float* __restrict__ src, unsigned short* __restrict__ dst, int n) {
  int base = blockIdx.x * 1024 + threadIdx.x;
#pragma unroll
  for (int e = 0; e < 4; ++e) {
    int i = base + e * 256;
    if (i < n) dst[i] = f2bf(src[i]);
  }
}

// ---------------- LayerNorm (rows of 1024) -> bf16 ----------------
__global__ __launch_bounds__(256) void ln_kernel(
    const float* __restrict__ x, const float* __restrict__ gw,
    const float* __restrict__ gb, unsigned short* __restrict__ out) {
  const int row = blockIdx.x;
  const int tid = threadIdx.x;
  const int wv = tid >> 5, l = tid & 31;
  __shared__ float red[16];
  const float* xr = x + (size_t)row * DDIM;
  float v[4];
  float s = 0.f, ss = 0.f;
#pragma unroll
  for (int e = 0; e < 4; ++e) {
    v[e] = xr[tid + 256 * e];
    s += v[e]; ss += v[e] * v[e];
  }
#pragma unroll
  for (int off = 16; off >= 1; off >>= 1) {
    s  += __shfl_xor(s,  off, 32);
    ss += __shfl_xor(ss, off, 32);
  }
  if (l == 0) { red[wv] = s; red[8 + wv] = ss; }
  __syncthreads();
  float S = 0.f, SS = 0.f;
#pragma unroll
  for (int i = 0; i < 8; ++i) { S += red[i]; SS += red[8 + i]; }
  float mu  = S * (1.0f / 1024.0f);
  float var = SS * (1.0f / 1024.0f) - mu * mu;
  float rs  = rsqrtf(var + 1e-5f);
#pragma unroll
  for (int e = 0; e < 4; ++e) {
    int c = tid + 256 * e;
    out[(size_t)row * DDIM + c] = f2bf((v[e] - mu) * rs * gw[c] + gb[c]);
  }
}

// ---------------- V transpose: qkv (v-part) -> vT[bh][dh][t] ----------------
__global__ __launch_bounds__(256) void transpose_v_kernel(
    const unsigned short* __restrict__ qkv, unsigned short* __restrict__ vT) {
  int idx = blockIdx.x * 256 + threadIdx.x;      // < 8*16*64*1024
  int t = idx & 1023;
  int rest = idx >> 10;
  int dh = rest & 63;
  int bh = rest >> 6;                            // b*16 + h
  int b = bh >> 4, h = bh & 15;
  unsigned short val = qkv[(size_t)(t * BB + b) * D3 + 2 * DDIM + h * 64 + dh];
  vT[(size_t)(bh * 64 + dh) * 1024 + t] = val;
}

// ---------------- Generic bf16 GEMM: C[M,N] = A[M,K] * B[N,K]^T -------------
// Double-buffered LDS pipeline fed by GLOBAL_LOAD_ASYNC_TO_LDS_B128 (ASYNCcnt).
// EPI: 0 = bf16 out; 1 = bias+ReLU -> bf16; 2 = +resid -> f32; 3 = bias+resid -> f32
template <int EPI>
__global__ __launch_bounds__(256) void gemm_bf16_kernel(
    const unsigned short* __restrict__ A, const unsigned short* __restrict__ B,
    int K, unsigned short* __restrict__ Cb, float* __restrict__ Cf,
    const float* __restrict__ bias, const float* __restrict__ resid, int N) {
  __shared__ unsigned short As[2][128 * 32];
  __shared__ unsigned short Bs[2][128 * 32];
  const int tid = threadIdx.x;
  const int wv = tid >> 5, l = tid & 31;
  const int lm = l & 15, lh = l >> 4;
  const int bm = blockIdx.x, bn = blockIdx.y;
  const int wm = (wv & 3) * 32;    // 2 m-tiles per wave
  const int wn = (wv >> 2) * 64;   // 4 n-tiles per wave
  const int ldr = tid >> 1;
  const int ldc = (tid & 1) * 16;
  v8f acc[2][4] = {};
  const unsigned short* ga0 = A + (size_t)(bm * 128 + ldr) * K + ldc;
  const unsigned short* gb0 = B + (size_t)(bn * 128 + ldr) * K + ldc;
  unsigned short* la0 = &As[0][ldr * 32 + ldc];
  unsigned short* lb0 = &Bs[0][ldr * 32 + ldc];
  const int lstride = 128 * 32;
  const int nsteps = K >> 5;

  // prologue: stage k-step 0 into buffer 0
  async_cp16(ga0,     la0);
  async_cp16(ga0 + 8, la0 + 8);
  async_cp16(gb0,     lb0);
  async_cp16(gb0 + 8, lb0 + 8);

  for (int it = 0; it < nsteps; ++it) {
    const int cur = it & 1;
    if (it + 1 < nsteps) {
      // stage next k-step into the other buffer (freed by barrier of it-1)
      const unsigned short* ga = ga0 + ((it + 1) << 5);
      const unsigned short* gb = gb0 + ((it + 1) << 5);
      unsigned short* la = la0 + (cur ^ 1) * lstride;
      unsigned short* lb = lb0 + (cur ^ 1) * lstride;
      async_cp16(ga,     la);
      async_cp16(ga + 8, la + 8);
      async_cp16(gb,     lb);
      async_cp16(gb + 8, lb + 8);
      asm volatile("s_wait_asynccnt 4" ::: "memory");  // current buffer arrived
    } else {
      asm volatile("s_wait_asynccnt 0" ::: "memory");
    }
    __syncthreads();  // every wave's async data visible to all

    Frag af[2], bfr[4];
#pragma unroll
    for (int mt = 0; mt < 2; ++mt) {
      const unsigned short* p = &As[cur][(wm + mt * 16 + lm) * 32];
      af[mt].q[0] = *(const uint4*)(p + lh * 16);
      af[mt].q[1] = *(const uint4*)(p + lh * 16 + 8);
    }
#pragma unroll
    for (int nt = 0; nt < 4; ++nt) {
      const unsigned short* p = &Bs[cur][(wn + nt * 16 + lm) * 32 + lh * 16];
      bfr[nt].q[0] = *(const uint4*)(p);
      bfr[nt].q[1] = *(const uint4*)(p + 8);
    }
#pragma unroll
    for (int mt = 0; mt < 2; ++mt)
#pragma unroll
      for (int nt = 0; nt < 4; ++nt)
        acc[mt][nt] = wmma_bf(af[mt], bfr[nt], acc[mt][nt]);
    __syncthreads();  // compute done before this buffer is refilled
  }
#pragma unroll
  for (int mt = 0; mt < 2; ++mt)
#pragma unroll
    for (int nt = 0; nt < 4; ++nt)
#pragma unroll
      for (int r = 0; r < 8; ++r) {
        int row = bm * 128 + wm + mt * 16 + r + lh * 8;
        int col = bn * 128 + wn + nt * 16 + lm;
        float v = acc[mt][nt][r];
        if (EPI == 1) { v += bias[col]; v = v > 0.f ? v : 0.f; }
        if (EPI == 3) { v += bias[col]; }
        if (EPI == 2 || EPI == 3) {
          v += resid[(size_t)row * N + col];
          Cf[(size_t)row * N + col] = v;
        } else {
          Cb[(size_t)row * N + col] = f2bf(v);
        }
      }
}

// ---------------- Fused relative attention ----------------
// grid: 2048 blocks = (b,h,qtile). LDS: S[64][1024] f32 + wqa[64][64] + wqb[80][64] bf16
__global__ __launch_bounds__(256) void attn_kernel(
    const unsigned short* __restrict__ qkv, const unsigned short* __restrict__ rk,
    const unsigned short* __restrict__ vT, const float* __restrict__ rwb,
    const float* __restrict__ rrb, unsigned short* __restrict__ attn_out) {
  extern __shared__ char smem[];
  float* S = (float*)smem;                                      // 64*1024 f32
  unsigned short* wqa = (unsigned short*)(smem + 64 * 1024 * 4);  // 64*64
  unsigned short* wqb = wqa + 64 * 64;                            // 80*64
  const int tid = threadIdx.x;
  const int wv = tid >> 5, l = tid & 31;
  const int lm = l & 15, lh = l >> 4;
  const int bid = blockIdx.x;
  const int it = bid & 15, h = (bid >> 4) & 15, b = bid >> 8;
  const int i0 = it * 64;

  // stage wq (+r_w_bias) and wq (+r_r_bias, 65 rows + zero pad) as bf16
#pragma unroll
  for (int e = 0; e < 16; ++e) {
    int idx = e * 256 + tid;
    int r = idx >> 6, d = idx & 63;
    int t = i0 + r;
    float v = bf2f(qkv[(size_t)(t * BB + b) * D3 + h * 64 + d]) + rwb[h * 64 + d];
    wqa[idx] = f2bf(v);
  }
#pragma unroll
  for (int e = 0; e < 20; ++e) {
    int idx = e * 256 + tid;
    int r = idx >> 6, d = idx & 63;
    int t = i0 + r;
    float v = 0.f;
    if (r <= 64 && t < TT)
      v = bf2f(qkv[(size_t)(t * BB + b) * D3 + h * 64 + d]) + rrb[h * 64 + d];
    wqb[idx] = f2bf(v);
  }
  __syncthreads();

  // ---- Phase A: AC = (wq + r_w_bias) @ wk^T -> S stripe ----
  {
    Frag af[4][2];
#pragma unroll
    for (int mt = 0; mt < 4; ++mt)
#pragma unroll
      for (int ks = 0; ks < 2; ++ks) {
        const unsigned short* p = wqa + (mt * 16 + lm) * 64 + ks * 32;
        af[mt][ks].q[0] = *(const uint4*)(p + lh * 16);
        af[mt][ks].q[1] = *(const uint4*)(p + lh * 16 + 8);
      }
    for (int nt = 0; nt < 8; ++nt) {
      int tcol = wv * 128 + nt * 16 + lm;
      Frag bfr[2];
#pragma unroll
      for (int ks = 0; ks < 2; ++ks) {
        const unsigned short* p =
            qkv + (size_t)(tcol * BB + b) * D3 + DDIM + h * 64 + ks * 32 + lh * 16;
        bfr[ks].q[0] = *(const uint4*)p;
        bfr[ks].q[1] = *(const uint4*)(p + 8);
      }
#pragma unroll
      for (int mt = 0; mt < 4; ++mt) {
        v8f acc = {};
        acc = wmma_bf(af[mt][0], bfr[0], acc);
        acc = wmma_bf(af[mt][1], bfr[1], acc);
#pragma unroll
        for (int r = 0; r < 8; ++r) {
          int m = mt * 16 + r + lh * 8;
          int col = wv * 128 + nt * 16 + lm;
          S[m * 1024 + col] = acc[r];
        }
      }
    }
  }
  __syncthreads();

  // ---- Phase B: BD = (wq + r_r_bias) @ r_k^T, rel_shift scatter-add into S ----
  {
    Frag af[5][2];
#pragma unroll
    for (int mt = 0; mt < 5; ++mt)
#pragma unroll
      for (int ks = 0; ks < 2; ++ks) {
        const unsigned short* p = wqb + (mt * 16 + lm) * 64 + ks * 32;
        af[mt][ks].q[0] = *(const uint4*)(p + lh * 16);
        af[mt][ks].q[1] = *(const uint4*)(p + lh * 16 + 8);
      }
    for (int nt = 0; nt < 8; ++nt) {
      int jj = wv * 128 + nt * 16 + lm;
      Frag bfr[2];
#pragma unroll
      for (int ks = 0; ks < 2; ++ks) {
        const unsigned short* p = rk + (size_t)jj * DDIM + h * 64 + ks * 32 + lh * 16;
        bfr[ks].q[0] = *(const uint4*)p;
        bfr[ks].q[1] = *(const uint4*)(p + 8);
      }
#pragma unroll
      for (int mt = 0; mt < 5; ++mt) {
        v8f acc = {};
        acc = wmma_bf(af[mt][0], bfr[0], acc);
        acc = wmma_bf(af[mt][1], bfr[1], acc);
#pragma unroll
        for (int r = 0; r < 8; ++r) {
          int ml = mt * 16 + r + lh * 8;   // local BD row (0..79)
          int rg = i0 + ml;                // global BD row
          float v = acc[r];
          int j1 = jj + rg - (TT - 1);
          if (j1 >= 0) {
            if (ml < 64) S[ml * 1024 + j1] += v;          // out[rg, j1], j1 <= rg
          } else if (ml >= 1 && ml <= 64) {
            S[(ml - 1) * 1024 + (j1 + TT)] += v;          // out[rg-1, jj+rg+1]
          }
        }
      }
    }
  }
  __syncthreads();

  // ---- Phase C: softmax over 1024 cols, repack probabilities bf16 in place ----
  {
    const float scale = 0.125f;  // 1/sqrt(64)
#pragma unroll 1
    for (int rr = 0; rr < 8; ++rr) {
      float* row = S + (wv * 8 + rr) * 1024;
      float mx = -3.4e38f;
#pragma unroll
      for (int k = 0; k < 32; ++k) mx = fmaxf(mx, row[l + 32 * k]);
#pragma unroll
      for (int off = 16; off >= 1; off >>= 1) mx = fmaxf(mx, __shfl_xor(mx, off, 32));
      float ev[32];
      float sum = 0.f;
#pragma unroll
      for (int k = 0; k < 32; ++k) {
        float e = __expf((row[l + 32 * k] - mx) * scale);
        ev[k] = e; sum += e;
      }
#pragma unroll
      for (int off = 16; off >= 1; off >>= 1) sum += __shfl_xor(sum, off, 32);
      float inv = 1.0f / sum;
      __threadfence_block();  // order f32 reads above vs bf16 writes below
      unsigned short* prow = (unsigned short*)row;
#pragma unroll
      for (int k = 0; k < 32; ++k) prow[l + 32 * k] = f2bf(ev[k] * inv);
    }
  }
  __syncthreads();

  // ---- Phase D: attn = P @ V  (P bf16 in LDS, V pre-transposed) ----
  {
    const unsigned short* Pb = (const unsigned short*)smem;  // row stride 2048 shorts
    const int mt = wv >> 1;
    const int ntb = (wv & 1) * 2;
    v8f acc0 = {}, acc1 = {};
    const unsigned short* vbase = vT + (size_t)((b * 16 + h) * 64) * 1024;
    for (int ks = 0; ks < 32; ++ks) {
      Frag af;
      const unsigned short* pa = Pb + (mt * 16 + lm) * 2048 + ks * 32;
      af.q[0] = *(const uint4*)(pa + lh * 16);
      af.q[1] = *(const uint4*)(pa + lh * 16 + 8);
      Frag bf0, bf1;
      const unsigned short* pv0 = vbase + (size_t)(ntb * 16 + lm) * 1024 + ks * 32 + lh * 16;
      bf0.q[0] = *(const uint4*)pv0;
      bf0.q[1] = *(const uint4*)(pv0 + 8);
      const unsigned short* pv1 = pv0 + 16 * 1024;
      bf1.q[0] = *(const uint4*)pv1;
      bf1.q[1] = *(const uint4*)(pv1 + 8);
      acc0 = wmma_bf(af, bf0, acc0);
      acc1 = wmma_bf(af, bf1, acc1);
    }
#pragma unroll
    for (int r = 0; r < 8; ++r) {
      int t = i0 + mt * 16 + r + lh * 8;
      int dh0 = ntb * 16 + lm;
      attn_out[(size_t)(t * BB + b) * DDIM + h * 64 + dh0]      = f2bf(acc0[r]);
      attn_out[(size_t)(t * BB + b) * DDIM + h * 64 + dh0 + 16] = f2bf(acc1[r]);
    }
  }
}

// ---------------- host launch ----------------
extern "C" void kernel_launch(void* const* d_in, const int* in_sizes, int n_in,
                              void* d_out, int out_size, void* d_ws, size_t ws_size,
                              hipStream_t stream) {
  const float* x1    = (const float*)d_in[0];
  const float* x2    = (const float*)d_in[1];
  const float* pos   = (const float*)d_in[2];
  const float* ln1w  = (const float*)d_in[3];
  const float* ln1b  = (const float*)d_in[4];
  const float* qkv_w = (const float*)d_in[5];
  const float* r_w   = (const float*)d_in[6];
  const float* rwb   = (const float*)d_in[7];
  const float* rrb   = (const float*)d_in[8];
  const float* o_w   = (const float*)d_in[9];
  const float* ln2w  = (const float*)d_in[10];
  const float* ln2b  = (const float*)d_in[11];
  const float* w1    = (const float*)d_in[12];
  const float* b1    = (const float*)d_in[13];
  const float* w2    = (const float*)d_in[14];
  const float* b2    = (const float*)d_in[15];

  uint8_t* ws = (uint8_t*)d_ws;
  unsigned short* q_in  = (unsigned short*)(ws + 0);            // 16 MB
  unsigned short* qkvb  = (unsigned short*)(ws + 16777216);     // 48 MB
  unsigned short* posb  = (unsigned short*)(ws + 67108864);     // 2 MB
  unsigned short* rkb   = (unsigned short*)(ws + 69206016);     // 2 MB
  unsigned short* vTb   = (unsigned short*)(ws + 71303168);     // 16 MB
  unsigned short* attnb = (unsigned short*)(ws + 88080384);     // 16 MB
  unsigned short* hinb  = (unsigned short*)(ws + 104857600);    // 16 MB
  unsigned short* ffhb  = (unsigned short*)(ws + 121634816);    // 64 MB
  unsigned short* qkvwb = (unsigned short*)(ws + 188743680);    // 6 MB
  unsigned short* rwwb  = (unsigned short*)(ws + 195035136);    // 2 MB
  unsigned short* owb   = (unsigned short*)(ws + 197132288);    // 2 MB
  unsigned short* w1b   = (unsigned short*)(ws + 199229440);    // 8 MB
  unsigned short* w2b   = (unsigned short*)(ws + 207618048);    // 8 MB

  float* y1 = (float*)d_out;
  float* y2 = y1 + (size_t)TT * BB * DDIM;

  // weight / pos conversions to bf16
  cvt_bf16_kernel<<<3072, 256, 0, stream>>>(qkv_w, qkvwb, D3 * DDIM);
  cvt_bf16_kernel<<<1024, 256, 0, stream>>>(r_w, rwwb, DDIM * DDIM);
  cvt_bf16_kernel<<<1024, 256, 0, stream>>>(o_w, owb, DDIM * DDIM);
  cvt_bf16_kernel<<<4096, 256, 0, stream>>>(w1, w1b, FF * DDIM);
  cvt_bf16_kernel<<<4096, 256, 0, stream>>>(w2, w2b, DDIM * FF);
  cvt_bf16_kernel<<<1024, 256, 0, stream>>>(pos, posb, TT * DDIM);

  // LN1(x2) -> q_in
  ln_kernel<<<TT * BB, 256, 0, stream>>>(x2, ln1w, ln1b, q_in);

  // qkv = q_in @ qkv_w^T   [8192 x 3072 x 1024]
  gemm_bf16_kernel<0><<<dim3(64, 24), 256, 0, stream>>>(
      q_in, qkvwb, DDIM, qkvb, nullptr, nullptr, nullptr, D3);

  // r_k = pos @ r_w^T      [1024 x 1024 x 1024]
  gemm_bf16_kernel<0><<<dim3(8, 8), 256, 0, stream>>>(
      posb, rwwb, DDIM, rkb, nullptr, nullptr, nullptr, DDIM);

  // transpose V for contiguous B-operand loads
  transpose_v_kernel<<<32768, 256, 0, stream>>>(qkvb, vTb);

  // fused relative attention
  const size_t smem_bytes = 64 * 1024 * 4 + 64 * 64 * 2 + 80 * 64 * 2;  // 280576
  attn_kernel<<<2048, 256, smem_bytes, stream>>>(qkvb, rkb, vTb, rwb, rrb, attnb);

  // y1 = attn @ o_w^T + x1  [8192 x 1024 x 1024] -> d_out (f32)
  gemm_bf16_kernel<2><<<dim3(64, 8), 256, 0, stream>>>(
      attnb, owb, DDIM, nullptr, y1, nullptr, x1, DDIM);

  // LN2(y1) -> hin
  ln_kernel<<<TT * BB, 256, 0, stream>>>(y1, ln2w, ln2b, hinb);

  // h = relu(hin @ w1^T + b1)  [8192 x 4096 x 1024] -> bf16
  gemm_bf16_kernel<1><<<dim3(64, 32), 256, 0, stream>>>(
      hinb, w1b, DDIM, ffhb, nullptr, b1, nullptr, FF);

  // y2 = h @ w2^T + b2 + x2  [8192 x 1024 x 4096] -> d_out second half (f32)
  gemm_bf16_kernel<3><<<dim3(64, 8), 256, 0, stream>>>(
      ffhb, w2b, FF, nullptr, y2, b2, x2, DDIM);
}
